// VisionImageTransformer_79852031967295
// MI455X (gfx1250) — matrix operand order repaired and checked by reading, not compiled, not run on hardware
//
#include <hip/hip_runtime.h>
#include <hip/hip_bf16.h>

// ---------------------------------------------------------------------------
// CDNA5 (gfx1250) ViT forward. All GEMMs + attention use
// v_wmma_f32_16x16x32_bf16 (wave32, fp32 accumulate). fp32 elsewhere.
// GEMM: 128x64 block tile, 8 waves (4x2), 32x32 per wave (4 acc),
// double-buffered LDS (one barrier per K-step) + global_prefetch_b8.
// ---------------------------------------------------------------------------

typedef __attribute__((ext_vector_type(16))) __bf16 v16bf;
typedef __attribute__((ext_vector_type(8)))  float  v8f;

union BFrag {
  v16bf v;
  unsigned int u[8];
  unsigned short h[16];
};

__device__ __forceinline__ unsigned short f2b(float f) {
  unsigned int u = __builtin_bit_cast(unsigned int, f);
  u += 0x7FFFu + ((u >> 16) & 1u);          // round-to-nearest-even
  return (unsigned short)(u >> 16);
}

// ------------------------- conv1 + relu + 2x2 maxpool ----------------------
// x: (16,512,512,3)  w: (3,3,3,16)  out: (16,255,255,16)
__global__ void conv1_pool_kernel(const float* __restrict__ x,
                                  const float* __restrict__ w,
                                  const float* __restrict__ bias,
                                  float* __restrict__ out) {
  int idx = blockIdx.x * blockDim.x + threadIdx.x;
  const int total = 16 * 255 * 255 * 16;
  if (idx >= total) return;
  int c  = idx & 15;
  int xo = (idx >> 4) % 255;
  int yo = (idx / (16 * 255)) % 255;
  int b  = idx / (16 * 255 * 255);
  float m = -3.402823466e38f;
  for (int dy = 0; dy < 2; ++dy)
    for (int dx = 0; dx < 2; ++dx) {
      int oy = 2 * yo + dy, ox = 2 * xo + dx;
      float acc = bias[c];
      for (int ky = 0; ky < 3; ++ky)
        for (int kx = 0; kx < 3; ++kx) {
          const float* xp = x + ((size_t)(b * 512 + oy + ky) * 512 + (ox + kx)) * 3;
          const float* wp = w + ((ky * 3 + kx) * 3) * 16 + c;
          acc += xp[0] * wp[0] + xp[1] * wp[16] + xp[2] * wp[32];
        }
      acc = fmaxf(acc, 0.f);
      m = fmaxf(m, acc);
    }
  out[idx] = m;
}

// ------------------------- conv2 + relu + 2x2 maxpool ----------------------
// in: (16,255,255,16)  w: (3,3,16,16)  out: (16,126,126,16)
__global__ void conv2_pool_kernel(const float* __restrict__ p1,
                                  const float* __restrict__ w,
                                  const float* __restrict__ bias,
                                  float* __restrict__ out) {
  int idx = blockIdx.x * blockDim.x + threadIdx.x;
  const int total = 16 * 126 * 126 * 16;
  if (idx >= total) return;
  int c  = idx & 15;
  int xo = (idx >> 4) % 126;
  int yo = (idx / (16 * 126)) % 126;
  int b  = idx / (16 * 126 * 126);
  float m = -3.402823466e38f;
  for (int dy = 0; dy < 2; ++dy)
    for (int dx = 0; dx < 2; ++dx) {
      int oy = 2 * yo + dy, ox = 2 * xo + dx;
      float acc = bias[c];
      for (int ky = 0; ky < 3; ++ky)
        for (int kx = 0; kx < 3; ++kx) {
          const float* xp = p1 + ((size_t)(b * 255 + oy + ky) * 255 + (ox + kx)) * 16;
          const float* wp = w + ((ky * 3 + kx) * 16) * 16 + c;
          #pragma unroll
          for (int ci = 0; ci < 16; ++ci) acc += xp[ci] * wp[ci * 16];
        }
      acc = fmaxf(acc, 0.f);
      m = fmaxf(m, acc);
    }
  out[idx] = m;
}

// ---------------- patch extraction (pad 126->128, 16x16 patches) -----------
// in: (16,126,126,16)  out bf16: (16,64,4096) == (1024 x 4096) row-major
__global__ void patch_kernel(const float* __restrict__ p2,
                             unsigned short* __restrict__ out) {
  int idx = blockIdx.x * blockDim.x + threadIdx.x;
  const int total = 16 * 64 * 4096;
  if (idx >= total) return;
  int i = idx & 4095;
  int p = (idx >> 12) & 63;
  int b = idx >> 18;
  int c = i & 15, lx = (i >> 4) & 15, ly = i >> 8;
  int py = p >> 3, px = p & 7;
  int y = py * 16 + ly - 1, x = px * 16 + lx - 1;
  float v = 0.f;
  if (y >= 0 && y < 126 && x >= 0 && x < 126)
    v = p2[((size_t)(b * 126 + y) * 126 + x) * 16 + c];
  out[idx] = f2b(v);
}

// --------------------------- f32 -> bf16 convert ---------------------------
__global__ void cvt_bf16_kernel(const float* __restrict__ in,
                                unsigned short* __restrict__ out, int n) {
  int i = blockIdx.x * blockDim.x + threadIdx.x;
  if (i < n) out[i] = f2b(in[i]);
}

// ------------------------------- WMMA GEMM ---------------------------------
// C = A(MxK,bf16) * W(KxN,bf16) + bias, optional ReLU; dual f32/bf16 store.
// 128x64 block tile; wave (wm 0..3, wn 0..1) owns 32x32; K-step 32;
// double-buffered LDS, one barrier per step.
__global__ __launch_bounds__(256) void gemm_bf16_kernel(
    const unsigned short* __restrict__ A, const unsigned short* __restrict__ Wt,
    const float* __restrict__ bias, float* __restrict__ Cf,
    unsigned short* __restrict__ Cb, int M, int N, int K, int relu) {
  __shared__ __align__(16) unsigned short As[2][128 * 32];  // [m][k]
  __shared__ __align__(16) unsigned short Bs[2][64 * 32];   // [n][k] transposed
  const int tid = threadIdx.x;
  const int lane = tid & 31, wave = tid >> 5;
  const int lm = lane & 15, half = lane >> 4;
  const int m0 = blockIdx.y * 128, n0 = blockIdx.x * 64;
  const int wm = wave & 3, wn = wave >> 2;
  v8f acc[2][2] = {};

  const int ar = tid >> 1, ac = (tid & 1) * 16;  // A staging: 128 rows x 32
  const int bk = tid >> 3, bn = (tid & 7) * 8;   // B staging: 32 k x 64 n

  // stage 0
  {
    const unsigned short* ga = &A[(size_t)(m0 + ar) * K + ac];
    *(uint4*)&As[0][ar * 32 + ac]     = *(const uint4*)(ga);
    *(uint4*)&As[0][ar * 32 + ac + 8] = *(const uint4*)(ga + 8);
    uint4 t = *(const uint4*)&Wt[(size_t)bk * N + n0 + bn];
    const unsigned short* th = (const unsigned short*)&t;
    #pragma unroll
    for (int j = 0; j < 8; ++j) Bs[0][(bn + j) * 32 + bk] = th[j];
  }
  __syncthreads();

  for (int kk = 0; kk < K; kk += 32) {
    const int cur = (kk >> 5) & 1, nxt = cur ^ 1;
    if (kk + 32 < K) {  // stage next while computing current
      const unsigned short* ga = &A[(size_t)(m0 + ar) * K + kk + 32 + ac];
      *(uint4*)&As[nxt][ar * 32 + ac]     = *(const uint4*)(ga);
      *(uint4*)&As[nxt][ar * 32 + ac + 8] = *(const uint4*)(ga + 8);
      uint4 t = *(const uint4*)&Wt[(size_t)(kk + 32 + bk) * N + n0 + bn];
      const unsigned short* th = (const unsigned short*)&t;
      #pragma unroll
      for (int j = 0; j < 8; ++j) Bs[nxt][(bn + j) * 32 + bk] = th[j];
      if (kk + 64 < K) {  // hide HBM latency for step after next
        __builtin_prefetch(&A[(size_t)(m0 + ar) * K + kk + 64 + ac], 0, 3);
        __builtin_prefetch(&Wt[(size_t)(kk + 64 + bk) * N + n0 + bn], 0, 3);
      }
    }
    BFrag a0, a1, b0, b1;
    #pragma unroll
    for (int vv = 0; vv < 8; ++vv) {
      int ka = half * 16 + vv * 2;                          // contiguous k, matches B
      a0.u[vv] = *(const unsigned int*)&As[cur][((wm * 2 + 0) * 16 + lm) * 32 + ka];
      a1.u[vv] = *(const unsigned int*)&As[cur][((wm * 2 + 1) * 16 + lm) * 32 + ka];
      int kb = half * 16 + vv * 2;                          // contiguous k, matches A
      b0.u[vv] = *(const unsigned int*)&Bs[cur][((wn * 2 + 0) * 16 + lm) * 32 + kb];
      b1.u[vv] = *(const unsigned int*)&Bs[cur][((wn * 2 + 1) * 16 + lm) * 32 + kb];
    }
    acc[0][0] = __builtin_amdgcn_wmma_f32_16x16x32_bf16(false, a0.v, false, b0.v,
                                                        (short)0, acc[0][0], false, false);
    acc[0][1] = __builtin_amdgcn_wmma_f32_16x16x32_bf16(false, a0.v, false, b1.v,
                                                        (short)0, acc[0][1], false, false);
    acc[1][0] = __builtin_amdgcn_wmma_f32_16x16x32_bf16(false, a1.v, false, b0.v,
                                                        (short)0, acc[1][0], false, false);
    acc[1][1] = __builtin_amdgcn_wmma_f32_16x16x32_bf16(false, a1.v, false, b1.v,
                                                        (short)0, acc[1][1], false, false);
    __syncthreads();
  }
  #pragma unroll
  for (int am = 0; am < 2; ++am) {
    #pragma unroll
    for (int tn = 0; tn < 2; ++tn) {
      int n = n0 + (wn * 2 + tn) * 16 + lm;
      float bval = bias ? bias[n] : 0.f;
      #pragma unroll
      for (int r = 0; r < 8; ++r) {
        int m = m0 + (wm * 2 + am) * 16 + half * 8 + r;   // ISA f32 C/D layout
        float v = acc[am][tn][r] + bval;
        if (relu) v = fmaxf(v, 0.f);
        if (Cf) Cf[(size_t)m * N + n] = v;
        if (Cb) Cb[(size_t)m * N + n] = f2b(v);
      }
    }
  }
}

// --------------------------- attention per (b,h) ---------------------------
// q,k,v f32 (1024x768) with head slice at col h*64; out bf16 flat (B,H,S,D)
// (flat (B,H,S,D) IS the reference's reshape(B,-1,E) memory layout).
__global__ __launch_bounds__(256) void attn_kernel(
    const float* __restrict__ Q, const float* __restrict__ Kk,
    const float* __restrict__ V, unsigned short* __restrict__ out) {
  __shared__ __align__(16) unsigned short qs[64 * 64];  // q, then probs
  __shared__ __align__(16) unsigned short ks[64 * 64];
  __shared__ __align__(16) unsigned short vs[64 * 64];
  __shared__ float sc[64 * 64];
  const int bh = blockIdx.x, b = bh / 12, h = bh % 12;
  const int tid = threadIdx.x;
  const int lane = tid & 31, wave = tid >> 5;
  const int lm = lane & 15, half = lane >> 4;

  {
    int s = tid >> 2, d0 = (tid & 3) * 16;
    size_t base = (size_t)(b * 64 + s) * 768 + h * 64 + d0;
    #pragma unroll
    for (int j = 0; j < 16; ++j) {
      qs[s * 64 + d0 + j] = f2b(Q[base + j]);
      ks[s * 64 + d0 + j] = f2b(Kk[base + j]);
      vs[s * 64 + d0 + j] = f2b(V[base + j]);
    }
  }
  __syncthreads();

  const int mi = wave & 3, nb = (wave >> 2) * 2;
  // S = q @ k^T * 0.125
  {
    v8f a0 = {}; v8f a1 = {};
    #pragma unroll
    for (int k0 = 0; k0 < 64; k0 += 32) {
      BFrag a, b0, b1;
      #pragma unroll
      for (int vv = 0; vv < 8; ++vv) {
        int ka = k0 + half * 16 + vv * 2;
        a.u[vv]  = *(const unsigned int*)&qs[(mi * 16 + lm) * 64 + ka];
        int kb = k0 + half * 16 + vv * 2;
        b0.u[vv] = *(const unsigned int*)&ks[((nb + 0) * 16 + lm) * 64 + kb];
        b1.u[vv] = *(const unsigned int*)&ks[((nb + 1) * 16 + lm) * 64 + kb];
      }
      a0 = __builtin_amdgcn_wmma_f32_16x16x32_bf16(false, a.v, false, b0.v,
                                                   (short)0, a0, false, false);
      a1 = __builtin_amdgcn_wmma_f32_16x16x32_bf16(false, a.v, false, b1.v,
                                                   (short)0, a1, false, false);
    }
    #pragma unroll
    for (int t = 0; t < 2; ++t) {
      v8f acc = t ? a1 : a0;
      int n = (nb + t) * 16 + lm;
      #pragma unroll
      for (int r = 0; r < 8; ++r)
        sc[(mi * 16 + half * 8 + r) * 64 + n] = acc[r] * 0.125f;
    }
  }
  __syncthreads();
  // softmax, probs -> qs (bf16)
  if (tid < 64) {
    float mx = -3.402823466e38f;
    for (int j = 0; j < 64; ++j) mx = fmaxf(mx, sc[tid * 64 + j]);
    float s = 0.f;
    for (int j = 0; j < 64; ++j) s += __expf(sc[tid * 64 + j] - mx);
    float inv = 1.f / s;
    for (int j = 0; j < 64; ++j)
      qs[tid * 64 + j] = f2b(__expf(sc[tid * 64 + j] - mx) * inv);
  }
  __syncthreads();
  // O = P @ V
  {
    v8f a0 = {}; v8f a1 = {};
    #pragma unroll
    for (int k0 = 0; k0 < 64; k0 += 32) {
      BFrag a, b0, b1;
      int n0c = (nb + 0) * 16 + lm, n1c = (nb + 1) * 16 + lm;
      #pragma unroll
      for (int vv = 0; vv < 8; ++vv) {
        int ka = k0 + half * 16 + vv * 2;
        a.u[vv] = *(const unsigned int*)&qs[(mi * 16 + lm) * 64 + ka];
        int kb = k0 + half * 16 + vv * 2;
        b0.h[2 * vv]     = vs[kb * 64 + n0c];
        b0.h[2 * vv + 1] = vs[(kb + 1) * 64 + n0c];
        b1.h[2 * vv]     = vs[kb * 64 + n1c];
        b1.h[2 * vv + 1] = vs[(kb + 1) * 64 + n1c];
      }
      a0 = __builtin_amdgcn_wmma_f32_16x16x32_bf16(false, a.v, false, b0.v,
                                                   (short)0, a0, false, false);
      a1 = __builtin_amdgcn_wmma_f32_16x16x32_bf16(false, a.v, false, b1.v,
                                                   (short)0, a1, false, false);
    }
    #pragma unroll
    for (int t = 0; t < 2; ++t) {
      v8f acc = t ? a1 : a0;
      int d = (nb + t) * 16 + lm;
      #pragma unroll
      for (int r = 0; r < 8; ++r) {
        int m = mi * 16 + half * 8 + r;
        out[((size_t)(b * 12 + h) * 64 + m) * 64 + d] = f2b(acc[r]);
      }
    }
  }
}

// -------------------- residual add + LayerNorm (per row) -------------------
__global__ __launch_bounds__(256) void ln_add_kernel(
    const float* __restrict__ A, const float* __restrict__ Bv,
    const float* __restrict__ g, const float* __restrict__ be,
    float* __restrict__ Of, unsigned short* __restrict__ Ob) {
  __shared__ float s1[256], s2[256];
  int row = blockIdx.x, tid = threadIdx.x;
  float sum = 0.f, sq = 0.f;
  for (int j = tid; j < 768; j += 256) {
    float v = A[(size_t)row * 768 + j] + Bv[(size_t)row * 768 + j];
    sum += v; sq += v * v;
  }
  s1[tid] = sum; s2[tid] = sq;
  __syncthreads();
  for (int st = 128; st > 0; st >>= 1) {
    if (tid < st) { s1[tid] += s1[tid + st]; s2[tid] += s2[tid + st]; }
    __syncthreads();
  }
  float mean = s1[0] * (1.f / 768.f);
  float var  = s2[0] * (1.f / 768.f) - mean * mean;
  float rstd = rsqrtf(var + 1e-6f);
  for (int j = tid; j < 768; j += 256) {
    float v = A[(size_t)row * 768 + j] + Bv[(size_t)row * 768 + j];
    float y = (v - mean) * rstd * g[j] + be[j];
    Of[(size_t)row * 768 + j] = y;
    Ob[(size_t)row * 768 + j] = f2b(y);
  }
}

// --------------------- mean pool + head + sigmoid --------------------------
__global__ __launch_bounds__(256) void head_kernel(
    const float* __restrict__ T, const float* __restrict__ hw,
    const float* __restrict__ hb, float* __restrict__ out) {
  __shared__ float red[256];
  int b = blockIdx.x, tid = threadIdx.x;
  float acc = 0.f;
  for (int e = tid; e < 768; e += 256) {
    float s = 0.f;
    for (int si = 0; si < 64; ++si) s += T[(size_t)(b * 64 + si) * 768 + e];
    acc += (s * (1.f / 64.f)) * hw[e];
  }
  red[tid] = acc;
  __syncthreads();
  for (int st = 128; st > 0; st >>= 1) {
    if (tid < st) red[tid] += red[tid + st];
    __syncthreads();
  }
  if (tid == 0) out[b] = 1.f / (1.f + __expf(-(red[0] + hb[0])));
}

// ---------------------------------------------------------------------------
extern "C" void kernel_launch(void* const* d_in, const int* in_sizes, int n_in,
                              void* d_out, int out_size, void* d_ws, size_t ws_size,
                              hipStream_t stream) {
  (void)in_sizes; (void)n_in; (void)out_size; (void)ws_size;
  const float* x     = (const float*)d_in[0];
  const float* c1w   = (const float*)d_in[1];
  const float* c1b   = (const float*)d_in[2];
  const float* c2w   = (const float*)d_in[3];
  const float* c2b   = (const float*)d_in[4];
  const float* projw = (const float*)d_in[5];
  const float* projb = (const float*)d_in[6];
  const float* Wq    = (const float*)d_in[7];
  const float* bq    = (const float*)d_in[8];
  const float* Wk    = (const float*)d_in[9];
  const float* bk    = (const float*)d_in[10];
  const float* Wv    = (const float*)d_in[11];
  const float* bv    = (const float*)d_in[12];
  const float* Wo    = (const float*)d_in[13];
  const float* bo    = (const float*)d_in[14];
  const float* W1    = (const float*)d_in[15];
  const float* b1    = (const float*)d_in[16];
  const float* W2    = (const float*)d_in[17];
  const float* b2    = (const float*)d_in[18];
  const float* ln1g  = (const float*)d_in[19];
  const float* ln1b  = (const float*)d_in[20];
  const float* ln2g  = (const float*)d_in[21];
  const float* ln2b  = (const float*)d_in[22];
  const float* hw    = (const float*)d_in[23];
  const float* hb    = (const float*)d_in[24];
  float* out = (float*)d_out;

  char* ws = (char*)d_ws;
  size_t off = 0;
  auto alloc = [&](size_t bytes) -> void* {
    void* p = ws + off;
    off = (off + bytes + 255) & ~(size_t)255;
    return p;
  };

  const size_t WE = 768 * 768, FE = (size_t)768 * 3072;
  unsigned short* projw_b = (unsigned short*)alloc((size_t)4096 * 768 * 2);
  unsigned short* wq_b = (unsigned short*)alloc(8 * WE * 2);
  unsigned short* wk_b = (unsigned short*)alloc(8 * WE * 2);
  unsigned short* wv_b = (unsigned short*)alloc(8 * WE * 2);
  unsigned short* wo_b = (unsigned short*)alloc(8 * WE * 2);
  unsigned short* w1_b = (unsigned short*)alloc(8 * FE * 2);
  unsigned short* w2_b = (unsigned short*)alloc(8 * FE * 2);
  float* pooled1 = (float*)alloc((size_t)16 * 255 * 255 * 16 * 4);
  float* pooled2 = (float*)alloc((size_t)16 * 126 * 126 * 16 * 4);
  unsigned short* patches = (unsigned short*)alloc((size_t)1024 * 4096 * 2);
  float* t_f  = (float*)alloc((size_t)1024 * 768 * 4);
  unsigned short* t_b = (unsigned short*)alloc((size_t)1024 * 768 * 2);
  float* q_f  = (float*)alloc((size_t)1024 * 768 * 4);
  float* k_f  = (float*)alloc((size_t)1024 * 768 * 4);
  float* v_f  = (float*)alloc((size_t)1024 * 768 * 4);
  unsigned short* attn_b = (unsigned short*)alloc((size_t)1024 * 768 * 2);
  float* o_f  = (float*)alloc((size_t)1024 * 768 * 4);
  float* o1_f = (float*)alloc((size_t)1024 * 768 * 4);
  unsigned short* o1_b = (unsigned short*)alloc((size_t)1024 * 768 * 2);
  unsigned short* ff_b = (unsigned short*)alloc((size_t)1024 * 3072 * 2);
  float* ff2_f = (float*)alloc((size_t)1024 * 768 * 4);

  auto cvt = [&](const float* src, unsigned short* dst, size_t n) {
    cvt_bf16_kernel<<<(unsigned)((n + 255) / 256), 256, 0, stream>>>(src, dst, (int)n);
  };
  cvt(projw, projw_b, (size_t)4096 * 768);
  cvt(Wq, wq_b, 8 * WE); cvt(Wk, wk_b, 8 * WE);
  cvt(Wv, wv_b, 8 * WE); cvt(Wo, wo_b, 8 * WE);
  cvt(W1, w1_b, 8 * FE); cvt(W2, w2_b, 8 * FE);

  conv1_pool_kernel<<<(16 * 255 * 255 * 16 + 255) / 256, 256, 0, stream>>>(x, c1w, c1b, pooled1);
  conv2_pool_kernel<<<(16 * 126 * 126 * 16 + 255) / 256, 256, 0, stream>>>(pooled1, c2w, c2b, pooled2);
  patch_kernel<<<(16 * 64 * 4096 + 255) / 256, 256, 0, stream>>>(pooled2, patches);

  // t = patches @ proj_w + proj_b   (1024 x 4096 x 768)
  gemm_bf16_kernel<<<dim3(768 / 64, 1024 / 128), 256, 0, stream>>>(
      patches, projw_b, projb, t_f, t_b, 1024, 768, 4096, 0);

  for (int i = 0; i < 8; ++i) {
    gemm_bf16_kernel<<<dim3(12, 8), 256, 0, stream>>>(
        t_b, wq_b + (size_t)i * WE, bq + i * 768, q_f, nullptr, 1024, 768, 768, 0);
    gemm_bf16_kernel<<<dim3(12, 8), 256, 0, stream>>>(
        t_b, wk_b + (size_t)i * WE, bk + i * 768, k_f, nullptr, 1024, 768, 768, 0);
    gemm_bf16_kernel<<<dim3(12, 8), 256, 0, stream>>>(
        t_b, wv_b + (size_t)i * WE, bv + i * 768, v_f, nullptr, 1024, 768, 768, 0);
    attn_kernel<<<16 * 12, 256, 0, stream>>>(q_f, k_f, v_f, attn_b);
    gemm_bf16_kernel<<<dim3(12, 8), 256, 0, stream>>>(
        attn_b, wo_b + (size_t)i * WE, bo + i * 768, o_f, nullptr, 1024, 768, 768, 0);
    ln_add_kernel<<<1024, 256, 0, stream>>>(
        t_f, o_f, ln1g + i * 768, ln1b + i * 768, o1_f, o1_b);
    gemm_bf16_kernel<<<dim3(3072 / 64, 8), 256, 0, stream>>>(
        o1_b, w1_b + (size_t)i * FE, b1 + i * 3072, nullptr, ff_b, 1024, 3072, 768, 1);
    gemm_bf16_kernel<<<dim3(12, 8), 256, 0, stream>>>(
        ff_b, w2_b + (size_t)i * FE, b2 + i * 768, ff2_f, nullptr, 1024, 768, 3072, 0);
    ln_add_kernel<<<1024, 256, 0, stream>>>(
        o1_f, ff2_f, ln2g + i * 768, ln2b + i * 768, t_f, t_b);
  }
  head_kernel<<<16, 256, 0, stream>>>(t_f, hw, hb, out);
}
